// MultiHeadAttention_5935644803340
// MI455X (gfx1250) — matrix operand order repaired and checked by reading, not compiled, not run on hardware
//
#include <hip/hip_runtime.h>

// ---------------------------------------------------------------------------
// Types for CDNA5 WMMA (wave32, 16x16x32 bf16 -> f32)
// ---------------------------------------------------------------------------
typedef __bf16 bf16_t;
typedef __attribute__((ext_vector_type(16))) __bf16 v16bf;
typedef __attribute__((ext_vector_type(8)))  __bf16 v8bf;
typedef __attribute__((ext_vector_type(4)))  __bf16 v4bf;
typedef __attribute__((ext_vector_type(8)))  float  v8f;
typedef __attribute__((ext_vector_type(4)))  float  v4f;
typedef __attribute__((ext_vector_type(2)))  float  v2f;
typedef int v4i __attribute__((vector_size(16)));   // matches builtin's V4i

#define WMMA_BF16(A_, B_, C_) \
    __builtin_amdgcn_wmma_f32_16x16x32_bf16(false, (A_), false, (B_), (short)0, (C_), false, false)

#if __has_builtin(__builtin_amdgcn_global_load_async_to_lds_b128)
#define HAVE_ASYNC_LDS 1
typedef __attribute__((address_space(1))) v4i* gv4i_p;
typedef __attribute__((address_space(3))) v4i* lv4i_p;
#else
#define HAVE_ASYNC_LDS 0
#endif

__device__ __forceinline__ void wait_async0() {
#if __has_builtin(__builtin_amdgcn_s_wait_asynccnt)
    __builtin_amdgcn_s_wait_asynccnt(0);
#else
    asm volatile("s_wait_asynccnt 0" ::: "memory");
#endif
}

__device__ __forceinline__ v16bf join16(v8bf lo, v8bf hi) {
    return __builtin_shufflevector(lo, hi, 0,1,2,3,4,5,6,7,8,9,10,11,12,13,14,15);
}
__device__ __forceinline__ v8f zero8() {
    v8f z;
#pragma unroll
    for (int i = 0; i < 8; ++i) z[i] = 0.0f;
    return z;
}
template <typename T>
__device__ __forceinline__ T nt_load(const T* p) { return __builtin_nontemporal_load(p); }

// Problem constants
static constexpr int Bc  = 16;
static constexpr int Sc  = 1024;
static constexpr int HIDc = 512;
static constexpr int NHc = 8;
static constexpr int DHc = 64;
static constexpr int DBc = 6;
static constexpr int Mc  = Bc * Sc;      // 16384 rows

// ---------------------------------------------------------------------------
// K0: transpose + convert a 512x512 f32 weight (W[k][n]) -> bf16 Wt[n][k]
// ---------------------------------------------------------------------------
__global__ __launch_bounds__(256)
void prep_weight(const float* __restrict__ W, bf16_t* __restrict__ Wt) {
    int idx = blockIdx.x * 256 + threadIdx.x;   // 512*512 / 256 = 1024 blocks
    int k = idx >> 9;
    int n = idx & 511;
    Wt[(size_t)n * 512 + k] = (bf16_t)W[(size_t)k * 512 + n];
}

// ---------------------------------------------------------------------------
// K1/K3: C[M,512] = A[M,512] @ W + bias.  Wt is bf16 [N=512][K=512] (pre-
// transposed).  A is f32 (read-once -> NT loads, converted to bf16 while
// staging to LDS) or bf16 (L2-resident ctx).
// Tile: 128x128x32, 8 waves, each wave computes 64x32 via 4x2 WMMA tiles.
// ---------------------------------------------------------------------------
template <bool A_F32, bool OUT_F32>
__global__ __launch_bounds__(256)
void gemm512(const void* __restrict__ Aptr, const bf16_t* __restrict__ Wt,
             const float* __restrict__ bias, void* __restrict__ Out) {
    constexpr int K = 512, N = 512;
    const int tid  = threadIdx.x;
    const int lane = tid & 31;
    const int wave = tid >> 5;
    const int hl   = lane >> 4;       // half-of-wave select
    const int l16  = lane & 15;
    const int wm   = wave & 1;        // 2 waves over M
    const int wn   = wave >> 1;       // 4 waves over N
    const int m0   = blockIdx.x * 128;
    const int n0   = blockIdx.y * 128;

    __shared__ __align__(16) bf16_t lds_a[128 * 40];  // 128 rows x 32 (+8 pad)
    __shared__ __align__(16) bf16_t lds_w[128 * 40];

    v8f acc[4][2];
#pragma unroll
    for (int mi = 0; mi < 4; ++mi)
#pragma unroll
        for (int ni = 0; ni < 2; ++ni) acc[mi][ni] = zero8();

    for (int k0 = 0; k0 < K; k0 += 32) {
        __syncthreads();
        // ---- stage A tile (convert f32 -> bf16 on the fly) ----
        if (A_F32) {
            const float* A = (const float*)Aptr;
#pragma unroll
            for (int i = 0; i < 4; ++i) {
                int c   = tid + i * 256;          // 1024 float4 chunks
                int row = c >> 3;
                int cp  = (c & 7) * 4;
                v4f f = nt_load((const v4f*)(A + (size_t)(m0 + row) * K + k0 + cp));
                v4bf h;
                h[0] = (bf16_t)f[0]; h[1] = (bf16_t)f[1];
                h[2] = (bf16_t)f[2]; h[3] = (bf16_t)f[3];
                *(v4bf*)(lds_a + row * 40 + cp) = h;
            }
        } else {
            const bf16_t* A = (const bf16_t*)Aptr;
#pragma unroll
            for (int i = 0; i < 2; ++i) {
                int c   = tid + i * 256;          // 512 v8bf chunks
                int row = c >> 2;
                int cp  = (c & 3) * 8;
                *(v8bf*)(lds_a + row * 40 + cp) =
                    *(const v8bf*)(A + (size_t)(m0 + row) * K + k0 + cp);
            }
        }
        // ---- stage W tile (already bf16, [n][k] layout) ----
#pragma unroll
        for (int i = 0; i < 2; ++i) {
            int c   = tid + i * 256;
            int row = c >> 2;
            int cp  = (c & 3) * 8;
            *(v8bf*)(lds_w + row * 40 + cp) =
                *(const v8bf*)(Wt + (size_t)(n0 + row) * K + k0 + cp);
        }
        __syncthreads();

        // ---- fragments (ISA VGPR layouts) ----
        v16bf af[4], bfr[2];
#pragma unroll
        for (int mi = 0; mi < 4; ++mi) {
            int r  = wm * 64 + mi * 16 + l16;
            int c0 = hl * 8;
            af[mi] = join16(*(const v8bf*)(lds_a + r * 40 + c0),
                            *(const v8bf*)(lds_a + r * 40 + c0 + 16));
        }
#pragma unroll
        for (int ni = 0; ni < 2; ++ni) {
            int r  = wn * 32 + ni * 16 + l16;
            int kb = hl * 8;
            bfr[ni] = join16(*(const v8bf*)(lds_w + r * 40 + kb),
                             *(const v8bf*)(lds_w + r * 40 + kb + 16));
        }
#pragma unroll
        for (int mi = 0; mi < 4; ++mi)
#pragma unroll
            for (int ni = 0; ni < 2; ++ni)
                acc[mi][ni] = WMMA_BF16(af[mi], bfr[ni], acc[mi][ni]);
    }

    // ---- epilogue: C/D layout row = r + 8*hl, col = l16 ----
#pragma unroll
    for (int ni = 0; ni < 2; ++ni) {
        int   col = n0 + wn * 32 + ni * 16 + l16;
        float bv  = bias[col];
#pragma unroll
        for (int mi = 0; mi < 4; ++mi) {
#pragma unroll
            for (int r = 0; r < 8; ++r) {
                int   row = m0 + wm * 64 + mi * 16 + r + 8 * hl;
                float val = acc[mi][ni][r] + bv;
                if (OUT_F32)
                    __builtin_nontemporal_store(val, (float*)Out + (size_t)row * N + col);
                else
                    ((bf16_t*)Out)[(size_t)row * N + col] = (bf16_t)val;
            }
        }
    }
}

// ---------------------------------------------------------------------------
// K2: fused bias-projection + flash attention.
// Grid: (B * NH * S/128) blocks, 256 threads.  Each wave owns 16 q-rows.
// j-loop over kv in tiles of 32; K/V tiles double-buffered in LDS, K tile
// staged with GLOBAL_LOAD_ASYNC_TO_LDS_B128 one tile ahead so the async DMA
// overlaps the WMMA + attn_bias stream.  attn_bias is read-once -> NT loads.
// ---------------------------------------------------------------------------
__global__ __launch_bounds__(256)
void attn_kernel(const bf16_t* __restrict__ qh, const bf16_t* __restrict__ kh,
                 const bf16_t* __restrict__ vh, const float* __restrict__ attn_bias,
                 const float* __restrict__ Wb, const float* __restrict__ bb,
                 bf16_t* __restrict__ ctx) {
    constexpr int S = Sc, HID = HIDc, NH = NHc, DH = DHc, DB = DBc;
    constexpr int KT = 32 * 72;   // one K tile   [kv=32][d=64 +pad]
    constexpr int VT = 64 * 40;   // one V^T tile [d=64][kv=32 +pad]
    const int tid  = threadIdx.x;
    const int lane = tid & 31;
    const int wave = tid >> 5;
    const int hl   = lane >> 4;
    const int l16  = lane & 15;

    const int blk  = blockIdx.x;
    const int qblk = blk & 7;             // S/128 = 8
    const int h    = (blk >> 3) & 7;      // NH = 8
    const int b    = blk >> 6;            // B = 16
    const int q0   = qblk * 128 + wave * 16;

    __shared__ __align__(16) bf16_t k_lds[2 * KT];      // double-buffered
    __shared__ __align__(16) bf16_t vt_lds[2 * VT];     // double-buffered
    __shared__ __align__(16) bf16_t p_lds[8 * 16 * 40]; // per-wave P scratch
    bf16_t* pw = p_lds + wave * (16 * 40);

    // Wb column for this head + bias
    float wb[6];
#pragma unroll
    for (int d = 0; d < 6; ++d) wb[d] = Wb[d * NH + h];
    const float bbh   = bb[h];
    const float scale = 0.125f;           // DH^-0.5

    // cooperative staging of one (K, V^T) tile pair into buffer `buf`
    const int kk = tid >> 3;              // kv row 0..31
    const int ch = (tid & 7) * 8;         // d chunk
    auto stage_tile = [&](int j0, int buf) {
        const bf16_t* gk = kh + (size_t)(b * S + j0 + kk) * HID + h * DH + ch;
        bf16_t* lk = k_lds + buf * KT + kk * 72 + ch;
#if HAVE_ASYNC_LDS
        __builtin_amdgcn_global_load_async_to_lds_b128((gv4i_p)gk, (lv4i_p)lk, 0, 0);
#else
        *(v8bf*)lk = *(const v8bf*)gk;
#endif
        v8bf vv = *(const v8bf*)(vh + (size_t)(b * S + j0 + kk) * HID + h * DH + ch);
        bf16_t* lv = vt_lds + buf * VT;
#pragma unroll
        for (int e = 0; e < 8; ++e) lv[(ch + e) * 40 + kk] = vv[e];
    };

    // Q A-fragments for this wave's 16 rows (d = 0..31 and 32..63)
    v16bf qa0, qa1;
    {
        const bf16_t* qrow = qh + (size_t)(b * S + q0 + l16) * HID + h * DH;
        int c0 = hl * 8;
        qa0 = join16(*(const v8bf*)(qrow + c0),      *(const v8bf*)(qrow + c0 + 16));
        qa1 = join16(*(const v8bf*)(qrow + 32 + c0), *(const v8bf*)(qrow + 32 + c0 + 16));
    }

    float m_r[8], l_r[8];
#pragma unroll
    for (int r = 0; r < 8; ++r) { m_r[r] = -1e30f; l_r[r] = 0.0f; }
    v8f o[4];
#pragma unroll
    for (int t = 0; t < 4; ++t) o[t] = zero8();

    const float* abase = attn_bias + (size_t)b * S * S * DB;

    stage_tile(0, 0);                     // prologue: stage first tile

    for (int jt = 0; jt < S / 32; ++jt) {
        const int j0  = jt * 32;
        const int buf = jt & 1;
        wait_async0();                    // this wave's async copies landed
        __syncthreads();                  // tile `buf` visible to all waves
        if (jt + 1 < S / 32)
            stage_tile(j0 + 32, buf ^ 1); // overlap next-tile DMA with compute

        const bf16_t* kbuf = k_lds + buf * KT;
        const bf16_t* vbuf = vt_lds + buf * VT;

        // ---- scores: S(16x32) = Q(16x64) . K^T(64x32) : 4 WMMAs ----
        v16bf b00, b01, b10, b11;
        {
            int kb = hl * 8;
            const bf16_t* r0 = kbuf + l16 * 72;          // kv cols 0..15
            const bf16_t* r1 = kbuf + (16 + l16) * 72;   // kv cols 16..31
            b00 = join16(*(const v8bf*)(r0 + kb),      *(const v8bf*)(r0 + kb + 16));
            b01 = join16(*(const v8bf*)(r0 + 32 + kb), *(const v8bf*)(r0 + 32 + kb + 16));
            b10 = join16(*(const v8bf*)(r1 + kb),      *(const v8bf*)(r1 + kb + 16));
            b11 = join16(*(const v8bf*)(r1 + 32 + kb), *(const v8bf*)(r1 + 32 + kb + 16));
        }
        v8f s0 = zero8(), s1 = zero8();
        s0 = WMMA_BF16(qa0, b00, s0);
        s0 = WMMA_BF16(qa1, b01, s0);
        s1 = WMMA_BF16(qa0, b10, s1);
        s1 = WMMA_BF16(qa1, b11, s1);

        // ---- fused pair-bias (NT stream) + online softmax (per row) ----
        float cf[8];
#pragma unroll
        for (int r = 0; r < 8; ++r) {
            int qi = q0 + r + 8 * hl;
            int c0 = j0 + l16, c1 = c0 + 16;
            const float* p0 = abase + ((size_t)qi * S + c0) * DB;
            const float* p1 = abase + ((size_t)qi * S + c1) * DB;
            float bias0 = bbh, bias1 = bbh;
#pragma unroll
            for (int d = 0; d < 6; d += 2) {          // 8B-aligned float2 NT loads
                v2f a0 = nt_load((const v2f*)(p0 + d));
                v2f a1 = nt_load((const v2f*)(p1 + d));
                bias0 += a0[0] * wb[d] + a0[1] * wb[d + 1];
                bias1 += a1[0] * wb[d] + a1[1] * wb[d + 1];
            }
            float e0 = s0[r] * scale + bias0;
            float e1 = s1[r] * scale + bias1;
            float mx = fmaxf(e0, e1);
#pragma unroll
            for (int off = 1; off < 16; off <<= 1)
                mx = fmaxf(mx, __shfl_xor(mx, off, 32));
            float mnew = fmaxf(m_r[r], mx);
            float pe0  = __expf(e0 - mnew);
            float pe1  = __expf(e1 - mnew);
            float ps   = pe0 + pe1;
#pragma unroll
            for (int off = 1; off < 16; off <<= 1)
                ps += __shfl_xor(ps, off, 32);
            float corr = __expf(m_r[r] - mnew);
            l_r[r] = l_r[r] * corr + ps;
            m_r[r] = mnew;
            cf[r]  = corr;
            int prow = r + 8 * hl;                    // re-layout P via LDS
            pw[prow * 40 + l16]      = (bf16_t)pe0;
            pw[prow * 40 + 16 + l16] = (bf16_t)pe1;
        }
        // same-wave LDS RAW: make sure stores land before A-frag reads
        asm volatile("s_wait_dscnt 0" ::: "memory");

        // rescale running output
#pragma unroll
        for (int t = 0; t < 4; ++t)
#pragma unroll
            for (int r = 0; r < 8; ++r) o[t][r] *= cf[r];

        // ---- O += P(16x32) . V(32x64) : 4 WMMAs ----
        v16bf pa;
        {
            int c0 = hl * 8;
            const bf16_t* pr = pw + l16 * 40;
            pa = join16(*(const v8bf*)(pr + c0), *(const v8bf*)(pr + c0 + 16));
        }
#pragma unroll
        for (int t = 0; t < 4; ++t) {
            const bf16_t* vr = vbuf + (t * 16 + l16) * 40 + hl * 8;
            v16bf vb = join16(*(const v8bf*)(vr), *(const v8bf*)(vr + 16));
            o[t] = WMMA_BF16(pa, vb, o[t]);
        }
    }

    // ---- normalize + store ctx (bf16, [b,s,h,d] flattened) ----
    float inv[8];
#pragma unroll
    for (int r = 0; r < 8; ++r) inv[r] = 1.0f / l_r[r];
#pragma unroll
    for (int t = 0; t < 4; ++t) {
#pragma unroll
        for (int r = 0; r < 8; ++r) {
            int row = q0 + r + 8 * hl;
            int col = h * DH + t * 16 + l16;
            ctx[(size_t)(b * S + row) * HID + col] = (bf16_t)(o[t][r] * inv[r]);
        }
    }
}

// ---------------------------------------------------------------------------
// Host-side orchestration
// ---------------------------------------------------------------------------
extern "C" void kernel_launch(void* const* d_in, const int* in_sizes, int n_in,
                              void* d_out, int out_size, void* d_ws, size_t ws_size,
                              hipStream_t stream) {
    (void)in_sizes; (void)n_in; (void)out_size; (void)ws_size;
    const float* q  = (const float*)d_in[0];
    const float* k  = (const float*)d_in[1];
    const float* v  = (const float*)d_in[2];
    const float* ab = (const float*)d_in[3];
    const float* Wq = (const float*)d_in[4];
    const float* bq = (const float*)d_in[5];
    const float* Wk = (const float*)d_in[6];
    const float* bk = (const float*)d_in[7];
    const float* Wv = (const float*)d_in[8];
    const float* bv = (const float*)d_in[9];
    const float* Wb = (const float*)d_in[10];
    const float* bb = (const float*)d_in[11];
    const float* Wo = (const float*)d_in[12];
    const float* bo = (const float*)d_in[13];
    float* out = (float*)d_out;

    char* ws = (char*)d_ws;
    const size_t SZ = (size_t)Mc * HIDc * sizeof(bf16_t);  // 16 MB each
    bf16_t* qh  = (bf16_t*)(ws);
    bf16_t* khp = (bf16_t*)(ws + SZ);
    bf16_t* vhp = (bf16_t*)(ws + 2 * SZ);
    bf16_t* ctx = (bf16_t*)(ws + 3 * SZ);
    bf16_t* Wqt = (bf16_t*)(ws + 4 * SZ);
    bf16_t* Wkt = Wqt + 512 * 512;
    bf16_t* Wvt = Wkt + 512 * 512;
    bf16_t* Wot = Wvt + 512 * 512;

    prep_weight<<<1024, 256, 0, stream>>>(Wq, Wqt);
    prep_weight<<<1024, 256, 0, stream>>>(Wk, Wkt);
    prep_weight<<<1024, 256, 0, stream>>>(Wv, Wvt);
    prep_weight<<<1024, 256, 0, stream>>>(Wo, Wot);

    dim3 g(Mc / 128, HIDc / 128);
    gemm512<true,  false><<<g, 256, 0, stream>>>(q, Wqt, bq, qh);
    gemm512<true,  false><<<g, 256, 0, stream>>>(k, Wkt, bk, khp);
    gemm512<true,  false><<<g, 256, 0, stream>>>(v, Wvt, bv, vhp);

    attn_kernel<<<Bc * NHc * (Sc / 128), 256, 0, stream>>>(qh, khp, vhp, ab, Wb, bb, ctx);

    gemm512<false, true><<<g, 256, 0, stream>>>(ctx, Wot, bo, out);
}
